// AC_RPPO_5617817223175
// MI455X (gfx1250) — compile-verified, matrix operand order checked by reading, not run on hardware
//
#include <hip/hip_runtime.h>

// ---------------------------------------------------------------------------
// MI455X (gfx1250) recurrent actor-critic forward pass.
//   - All GEMM-shaped math via v_wmma_f32_16x16x32_bf16 (bf16 in, f32 accum).
//   - Convs: implicit GEMM; bias + 2x2 maxpool (shfl_xor 1/8 inside the D
//     fragment) + ReLU fused in the epilogue -> no pre-pool HBM traffic.
//   - conv3/conv4: whole input image staged into LDS via the Tensor Data
//     Mover (tensor_load_to_lds + s_wait_tensorcnt), im2col gathers hit LDS.
//   - LSTM: one 32-wave workgroup on a WGP; h (64x512 bf16) in LDS, c in
//     registers; per step 512 WMMA tiles (K=1024 input + K=512 recurrent).
//   - Inter-layer activations (~155 MB bf16) stay resident in the 192 MB L2.
// ---------------------------------------------------------------------------

typedef __attribute__((ext_vector_type(16))) __bf16 v16bf;
typedef __attribute__((ext_vector_type(8)))  float  v8f;

#define DEVINL __device__ __forceinline__

DEVINL float sigmoidf_(float x) { return 1.0f / (1.0f + __expf(-x)); }

// A/B fragment k-map (ISA 16-bit 16x32 A layout):
// lane half hi: element e (0..7) -> k = hi*8 + e ; e (8..15) -> k = hi*8 + 16 + (e-8)
DEVINL v16bf load_frag(const __bf16* p, int hi) {
    v16bf r;
    const __bf16* q = p + hi * 8;
#pragma unroll
    for (int e = 0; e < 8; ++e) { r[e] = q[e]; r[e + 8] = q[e + 16]; }
    return r;
}

DEVINL v8f wmma_bf16(v16bf a, v16bf b, v8f c) {
    // (neg_a, A, neg_b, B, c_mod, C, reuse_a, reuse_b)
    return __builtin_amdgcn_wmma_f32_16x16x32_bf16(false, a, false, b, (short)0, c,
                                                   false, false);
}

// ---------------------------------------------------------------------------
// Tensor Data Mover: 1-D global -> LDS copy descriptor (ISA cdna5 ch.8).
// ---------------------------------------------------------------------------
#if defined(__AMDGCN__) && __has_builtin(__builtin_amdgcn_tensor_load_to_lds)
#define HAVE_TDM 1
#else
#define HAVE_TDM 0
#endif

#if HAVE_TDM
typedef __attribute__((ext_vector_type(4))) unsigned int u32x4_;
typedef __attribute__((ext_vector_type(8))) int i32x8_;
typedef __attribute__((ext_vector_type(4))) int i32x4_;

// Copy ndwords DWORDs from gptr (global) to LDS byte offset lds_off.
// D# group0: count=1 | lds_addr | global_addr(57b) | type=2.
// D# group1: data_size=4B, tensor_dim0=tile_dim0=ndwords (1-D tile), stride=ndwords.
DEVINL void tdm_load_1d(unsigned lds_off, const void* gptr, unsigned ndwords) {
    unsigned long long ga = (unsigned long long)gptr;
    u32x4_ g0;
    g0[0] = 1u;                                           // count=1 (valid user D#)
    g0[1] = lds_off;                                      // lds_addr (bytes)
    g0[2] = (unsigned)(ga & 0xffffffffu);                 // global_addr[31:0]
    g0[3] = (unsigned)((ga >> 32) & 0x1ffffffu)           // global_addr[56:32]
            | (2u << 30);                                 // type=2 ("image")
    i32x8_ g1;
    g1[0] = (int)(2u << 16);                              // data_size=2 -> 4 bytes
    g1[1] = (int)((ndwords & 0xffffu) << 16);             // tensor_dim0[15:0]
    g1[2] = (int)(((ndwords >> 16) & 0xffffu)             // tensor_dim0[31:16]
            | (1u << 16));                                // tensor_dim1 = 1
    g1[3] = (int)((ndwords & 0xffffu) << 16);             // tile_dim0 (<=65535)
    g1[4] = 0;                                            // tile_dim1/2 unused
    g1[5] = (int)ndwords;                                 // tensor_dim0_stride lo
    g1[6] = 0;
    g1[7] = 0;
    i32x4_ z4 = {0, 0, 0, 0};
#if defined(__clang_major__) && __clang_major__ >= 23
    i32x8_ z8 = {0, 0, 0, 0, 0, 0, 0, 0};
    __builtin_amdgcn_tensor_load_to_lds(g0, g1, z4, z4, z8, 0);
#else
    __builtin_amdgcn_tensor_load_to_lds(g0, g1, z4, z4, 0);
#endif
}
#endif

// ---------------------------------------------------------------------------
// Prep kernels
// ---------------------------------------------------------------------------

__global__ void k_cvt_bf16(const float* __restrict__ s, __bf16* __restrict__ d, int n) {
    for (int i = blockIdx.x * blockDim.x + threadIdx.x; i < n; i += gridDim.x * blockDim.x)
        d[i] = (__bf16)s[i];
}

__global__ void k_bias_sum(const float* __restrict__ a, const float* __restrict__ b,
                           float* __restrict__ o) {
    int i = blockIdx.x * blockDim.x + threadIdx.x;
    if (i < 2048) o[i] = a[i] + b[i];
}

// Pack actor (18x512) + critic (1x512) into 32x512 bf16 (rows 19..31 zero).
__global__ void k_pack_head(const float* __restrict__ Wa, const float* __restrict__ Wc,
                            __bf16* __restrict__ hw) {
    int idx = blockIdx.x * blockDim.x + threadIdx.x;
    if (idx >= 32 * 512) return;
    int row = idx >> 9, k = idx & 511;
    float v = 0.f;
    if (row < 18)       v = Wa[row * 512 + k];
    else if (row == 18) v = Wc[k];
    hw[idx] = (__bf16)v;
}

// Pack OIHW conv weights into per-(mtile,kstep,lane) A fragments, per-channel-group
// k padded to 32 so every WMMA k-step is dense with zeros in the pad slots.
template <int IC, int KH, int KW, int OC, int CPK, int NSTEPS, int MT>
__global__ void k_pack_conv_w(const float* __restrict__ w, __bf16* __restrict__ apack) {
    int idx = blockIdx.x * blockDim.x + threadIdx.x;
    constexpr int TOTAL = MT * NSTEPS * 32 * 16;
    if (idx >= TOTAL) return;
    int e    = idx & 15;
    int lane = (idx >> 4) & 31;
    int s    = (idx >> 9) % NSTEPS;
    int mt   = (idx >> 9) / NSTEPS;
    int hi   = lane >> 4;
    int oc   = mt * 16 + (lane & 15);
    int k    = hi * 8 + (e & 7) + ((e >> 3) << 4);   // fragment k-map
    constexpr int KSZ = KH * KW;
    int coff = k / KSZ;
    int t    = k - coff * KSZ;
    int ic   = s * CPK + coff;
    float v = 0.f;
    if (oc < OC && coff < CPK && ic < IC)
        v = w[(((size_t)oc * IC + ic) * KH + (t / KW)) * KW + (t % KW)];
    apack[idx] = (__bf16)v;
}

// ---------------------------------------------------------------------------
// Shared conv tile body: computes one 16(oc) x 16(pos) tile where the 16
// positions form a pool-aligned 2x8 conv patch; epilogue does bias + 2x2
// cross-lane max + ReLU + bf16 store.  Gather source is templated (global/LDS).
// ---------------------------------------------------------------------------
template <typename SrcT, int IC, int IH, int IW, int KH, int KW, int PAD,
          int OC, int PH, int PW, int CPK, int NSTEPS, int MT, int TW>
DEVINL void conv_tile_body(const SrcT* src, const __bf16* __restrict__ apack,
                           const float* __restrict__ bias, __bf16* __restrict__ act,
                           int img, int tile, int lane) {
    constexpr int KSZ = KH * KW;
    const int p  = lane & 15;
    const int hi = lane >> 4;
    int mt  = tile % MT;
    int t2  = tile / MT;
    int tw  = t2 % TW;
    int thr = t2 / TW;                 // pool-output row
    int oh  = thr * 2 + (p >> 3);      // conv row for this lane
    int ow  = tw * 8 + (p & 7);        // conv col for this lane

    v8f acc = {};
    const __bf16* ap = apack + ((size_t)(mt * NSTEPS) * 32 + lane) * 16;
#pragma unroll 2
    for (int s = 0; s < NSTEPS; ++s) {
        v16bf a = *(const v16bf*)ap;
        ap += 32 * 16;
        v16bf b;
        int ic0 = s * CPK;
#pragma unroll
        for (int e = 0; e < 16; ++e) {
            int k    = hi * 8 + (e & 7) + ((e >> 3) << 4);
            int coff = k / KSZ;
            int t    = k - coff * KSZ;
            int ic   = ic0 + coff;
            int kh   = t / KW;
            int kw2  = t - kh * KW;
            int ih   = oh - PAD + kh;
            int iw   = ow - PAD + kw2;
            float v = 0.f;
            if (coff < CPK && ic < IC && ih >= 0 && ih < IH && iw >= 0 && iw < IW)
                v = (float)src[((size_t)ic * IH + ih) * IW + iw];
            b[e] = (__bf16)v;
        }
        acc = wmma_bf16(a, b, acc);
    }
    // bias -> 2x2 max over lanes {p, p^1, p^8, p^9} -> ReLU -> bf16 store
#pragma unroll
    for (int r = 0; r < 8; ++r) {
        int oc = mt * 16 + r + hi * 8;
        float v  = acc[r] + bias[oc];
        float v1 = fmaxf(v,  __shfl_xor(v,  1, 32));
        float v2 = fmaxf(v1, __shfl_xor(v1, 8, 32));
        if ((p & 1) == 0 && (p & 8) == 0) {
            int pw = tw * 4 + (p >> 1);
            if (pw < PW) {
                float o = fmaxf(v2, 0.f);
                act[(((size_t)img * OC + oc) * PH + thr) * PW + pw] = (__bf16)o;
            }
        }
    }
}

// Global-gather variant (conv1: f32 src, conv2: bf16 src; both L2-resident).
template <typename Tin, int IC, int IH, int IW, int KH, int KW, int PAD,
          int OC, int PH, int PW, int CPK, int NSTEPS, int MT, int TW>
__global__ __launch_bounds__(128) void conv_pool_wmma(
    const Tin* __restrict__ in, const __bf16* __restrict__ apack,
    const float* __restrict__ bias, __bf16* __restrict__ act) {
    const int img   = blockIdx.x;
    const int lane  = threadIdx.x & 31;
    const int wstrm = blockIdx.y * (blockDim.x >> 5) + (threadIdx.x >> 5);
    const int nstrm = gridDim.y * (blockDim.x >> 5);
    const Tin* src = in + (size_t)img * IC * IH * IW;
    __builtin_prefetch((const void*)src, 0, 1);
    constexpr int TOTAL = PH * TW * MT;
    for (int tile = wstrm; tile < TOTAL; tile += nstrm)
        conv_tile_body<Tin, IC, IH, IW, KH, KW, PAD, OC, PH, PW, CPK, NSTEPS, MT, TW>(
            src, apack, bias, act, img, tile, lane);
}

// LDS-staged variant (conv3/conv4: whole image fits in LDS; staged via TDM).
template <int IC, int IH, int IW, int KH, int KW, int PAD,
          int OC, int PH, int PW, int CPK, int NSTEPS, int MT, int TW>
__global__ __launch_bounds__(256) void conv_pool_wmma_lds(
    const __bf16* __restrict__ in, const __bf16* __restrict__ apack,
    const float* __restrict__ bias, __bf16* __restrict__ act) {
    constexpr int IMG_ELEMS = IC * IH * IW;            // even; *2B = mult of 4
    __shared__ __bf16 simg[IMG_ELEMS];                 // sole LDS object -> offset 0
    const int img = blockIdx.x;
    const __bf16* src = in + (size_t)img * IMG_ELEMS;
#if HAVE_TDM
    if (threadIdx.x == 0) {
        tdm_load_1d(0u, (const void*)src, (unsigned)(IMG_ELEMS * 2 / 4));
        __builtin_amdgcn_s_wait_tensorcnt(0);
    }
#else
    for (int i = threadIdx.x; i < IMG_ELEMS / 2; i += blockDim.x)
        ((unsigned*)simg)[i] = ((const unsigned*)src)[i];
#endif
    __syncthreads();

    const int lane  = threadIdx.x & 31;
    const int wstrm = threadIdx.x >> 5;
    const int nstrm = blockDim.x >> 5;
    constexpr int TOTAL = PH * TW * MT;
    for (int tile = wstrm; tile < TOTAL; tile += nstrm)
        conv_tile_body<__bf16, IC, IH, IW, KH, KW, PAD, OC, PH, PW, CPK, NSTEPS, MT, TW>(
            simg, apack, bias, act, img, tile, lane);
}

// ---------------------------------------------------------------------------
// LSTM: one workgroup (32 waves on one WGP). h (64x512 bf16) in LDS, c in
// registers (32 f32/thread). Per step: 512 WMMA tiles (128 gate-tiles x 4
// batch-tiles), K = 1024 (input) + 512 (recurrent), then gate nonlinearity.
// ---------------------------------------------------------------------------
__global__ __launch_bounds__(1024) void k_lstm(
    const __bf16* __restrict__ feat,   // (1024, 1024) = (T*B, FEAT)
    const float* __restrict__ done,    // (1024,)
    const float* __restrict__ h0, const float* __restrict__ c0,
    const __bf16* __restrict__ Wih,    // (2048, 1024) bf16 row-major
    const __bf16* __restrict__ Whh,    // (2048, 512)
    const float* __restrict__ biasSum, // (2048,)
    float* __restrict__ G,             // (2048, 64) scratch
    __bf16* __restrict__ hs,           // (1024, 512) all-step hidden, bf16
    float* __restrict__ outHT, float* __restrict__ outCT) {
    __shared__ __bf16 hbuf[64][512];   // 64 KB of the 320 KB WGP LDS

    const int tid  = threadIdx.x;
    const int lane = tid & 31;
    const int wave = tid >> 5;
    const int myb  = tid >> 4;         // batch row owned in elementwise phase
    const int hb   = (tid & 15) * 32;  // hid base owned
    const int p    = lane & 15;
    const int hi   = lane >> 4;

    float c[32];
#pragma unroll
    for (int j = 0; j < 32; ++j) {
        c[j] = c0[myb * 512 + hb + j];
        hbuf[myb][hb + j] = (__bf16)h0[myb * 512 + hb + j];
    }
    __syncthreads();

    for (int t = 0; t < 16; ++t) {
        // episode-reset mask: h *= m, c *= m
        float m = 1.0f - done[t * 64 + myb];
#pragma unroll
        for (int j = 0; j < 32; ++j) {
            c[j] *= m;
            hbuf[myb][hb + j] = (__bf16)((float)hbuf[myb][hb + j] * m);
        }
        __syncthreads();

        // G = Wih @ x_t^T + Whh @ h^T + bias   (G: gates x batch)
        for (int i = 0; i < 16; ++i) {
            int tileid = wave * 16 + i;
            int g0 = (tileid >> 2) * 16;
            int b0 = (tileid & 3) * 16;
            v8f acc;
#pragma unroll
            for (int r = 0; r < 8; ++r) acc[r] = biasSum[g0 + r + hi * 8];

            const __bf16* arow = Wih + (size_t)(g0 + p) * 1024;
            const __bf16* brow = feat + (size_t)(t * 64 + b0 + p) * 1024;
#pragma unroll 4
            for (int ks = 0; ks < 32; ++ks) {
                v16bf a = load_frag(arow + ks * 32, hi);
                v16bf b = load_frag(brow + ks * 32, hi);
                acc = wmma_bf16(a, b, acc);
            }
            const __bf16* arow2 = Whh + (size_t)(g0 + p) * 512;
            const __bf16* brow2 = &hbuf[b0 + p][0];
#pragma unroll 4
            for (int ks = 0; ks < 16; ++ks) {
                v16bf a = load_frag(arow2 + ks * 32, hi);
                v16bf b = load_frag(brow2 + ks * 32, hi);   // ds_load from LDS
                acc = wmma_bf16(a, b, acc);
            }
#pragma unroll
            for (int r = 0; r < 8; ++r)
                G[(size_t)(g0 + r + hi * 8) * 64 + b0 + p] = acc[r];
        }
        __threadfence();
        __syncthreads();

        // gate nonlinearity; update c (regs) and h (LDS + global bf16)
#pragma unroll 4
        for (int j = 0; j < 32; ++j) {
            int hid  = hb + j;
            float ig = G[(size_t)(hid)        * 64 + myb];
            float fg = G[(size_t)(512 + hid)  * 64 + myb];
            float gg = G[(size_t)(1024 + hid) * 64 + myb];
            float og = G[(size_t)(1536 + hid) * 64 + myb];
            float cn = sigmoidf_(fg) * c[j] + sigmoidf_(ig) * tanhf(gg);
            float hn = sigmoidf_(og) * tanhf(cn);
            c[j] = cn;
            hbuf[myb][hid] = (__bf16)hn;
            hs[(size_t)(t * 64 + myb) * 512 + hid] = (__bf16)hn;
            if (t == 15) {
                outHT[myb * 512 + hid] = hn;
                outCT[myb * 512 + hid] = cn;
            }
        }
        __syncthreads();
    }
}

// ---------------------------------------------------------------------------
// Heads: out(1024,19) = hidden(1024,512) @ [Wa;Wc]^T + [ba;bc] via WMMA.
// ---------------------------------------------------------------------------
__global__ __launch_bounds__(64) void k_head(
    const __bf16* __restrict__ hsb, const __bf16* __restrict__ hw,
    const float* __restrict__ ba, const float* __restrict__ bc,
    float* __restrict__ out) {
    const int lane = threadIdx.x & 31;
    const int mt   = threadIdx.x >> 5;   // 2 waves: output rows 0-15 / 16-31
    const int n0   = blockIdx.x * 16;
    const int p    = lane & 15;
    const int hi   = lane >> 4;
    v8f acc = {};
    const __bf16* arow = hw + (size_t)(mt * 16 + p) * 512;
    const __bf16* brow = hsb + (size_t)(n0 + p) * 512;
#pragma unroll 4
    for (int ks = 0; ks < 16; ++ks) {
        v16bf a = load_frag(arow + ks * 32, hi);
        v16bf b = load_frag(brow + ks * 32, hi);
        acc = wmma_bf16(a, b, acc);
    }
#pragma unroll
    for (int r = 0; r < 8; ++r) {
        int j = mt * 16 + r + hi * 8;
        if (j < 19) {
            float bias = (j < 18) ? ba[j] : bc[0];
            out[(size_t)(n0 + p) * 19 + j] = acc[r] + bias;
        }
    }
}

// ---------------------------------------------------------------------------
// Launcher
// ---------------------------------------------------------------------------
extern "C" void kernel_launch(void* const* d_in, const int* in_sizes, int n_in,
                              void* d_out, int out_size, void* d_ws, size_t ws_size,
                              hipStream_t stream) {
    (void)in_sizes; (void)n_in; (void)out_size; (void)ws_size;
    const float* x    = (const float*)d_in[0];
    const float* done = (const float*)d_in[1];
    const float* h0   = (const float*)d_in[2];
    const float* c0   = (const float*)d_in[3];
    const float* w1   = (const float*)d_in[4];
    const float* b1   = (const float*)d_in[5];
    const float* w2   = (const float*)d_in[6];
    const float* b2   = (const float*)d_in[7];
    const float* w3   = (const float*)d_in[8];
    const float* b3   = (const float*)d_in[9];
    const float* w4   = (const float*)d_in[10];
    const float* b4   = (const float*)d_in[11];
    const float* Wih  = (const float*)d_in[12];
    const float* Whh  = (const float*)d_in[13];
    const float* bih  = (const float*)d_in[14];
    const float* bhh  = (const float*)d_in[15];
    const float* Wa   = (const float*)d_in[16];
    const float* ba   = (const float*)d_in[17];
    const float* Wc   = (const float*)d_in[18];
    const float* bc   = (const float*)d_in[19];

    char* wp = (char*)d_ws;
    auto carve = [&](size_t elems, size_t esz) {
        char* pp = wp;
        wp += (elems * esz + 255) & ~(size_t)255;
        return pp;
    };
    __bf16* act1  = (__bf16*)carve(57802752, 2);  // (1024,32,42,42)
    __bf16* act2  = (__bf16*)carve(13107200, 2);  // (1024,32,20,20)
    __bf16* act3  = (__bf16*)carve(5308416, 2);   // (1024,64,9,9)
    __bf16* feat  = (__bf16*)carve(1048576, 2);   // (1024,64,4,4) = (T*B,1024)
    __bf16* ap1   = (__bf16*)carve(4096, 2);
    __bf16* ap2   = (__bf16*)carve(32768, 2);
    __bf16* ap3   = (__bf16*)carve(32768, 2);
    __bf16* ap4   = (__bf16*)carve(45056, 2);
    __bf16* wihb  = (__bf16*)carve(2097152, 2);
    __bf16* whhb  = (__bf16*)carve(1048576, 2);
    float*  bsum  = (float*) carve(2048, 4);
    __bf16* headw = (__bf16*)carve(16384, 2);
    float*  G     = (float*) carve(131072, 4);
    __bf16* hsbuf = (__bf16*)carve(524288, 2);

    float* out   = (float*)d_out;
    float* outHT = out + 1024 * 19;            // 19456
    float* outCT = outHT + 64 * 512;           // 52224

    // --- prep (weight conversion / fragment packing) ---
    k_cvt_bf16<<<4096, 256, 0, stream>>>(Wih, wihb, 2097152);
    k_cvt_bf16<<<2048, 256, 0, stream>>>(Whh, whhb, 1048576);
    k_bias_sum<<<8, 256, 0, stream>>>(bih, bhh, bsum);
    k_pack_head<<<64, 256, 0, stream>>>(Wa, Wc, headw);
    k_pack_conv_w<4, 5, 5, 32, 1, 4, 2><<<16, 256, 0, stream>>>(w1, ap1);
    k_pack_conv_w<32, 5, 5, 32, 1, 32, 2><<<128, 256, 0, stream>>>(w2, ap2);
    k_pack_conv_w<32, 4, 4, 64, 2, 16, 4><<<128, 256, 0, stream>>>(w3, ap3);
    k_pack_conv_w<64, 3, 3, 64, 3, 22, 4><<<176, 256, 0, stream>>>(w4, ap4);

    // --- backbone: conv+bias+pool+relu, all WMMA ---
    conv_pool_wmma<float, 4, 84, 84, 5, 5, 2, 32, 42, 42, 1, 4, 2, 11>
        <<<dim3(1024, 8), 128, 0, stream>>>(x, ap1, b1, act1);
    conv_pool_wmma<__bf16, 32, 42, 42, 5, 5, 1, 32, 20, 20, 1, 32, 2, 5>
        <<<dim3(1024, 4), 128, 0, stream>>>(act1, ap2, b2, act2);
    conv_pool_wmma_lds<32, 20, 20, 4, 4, 1, 64, 9, 9, 2, 16, 4, 3>
        <<<dim3(1024, 1), 256, 0, stream>>>(act2, ap3, b3, act3);
    conv_pool_wmma_lds<64, 9, 9, 3, 3, 1, 64, 4, 4, 3, 22, 4, 1>
        <<<dim3(1024, 1), 128, 0, stream>>>(act3, ap4, b4, feat);

    // --- recurrence (single workgroup, 32 waves) ---
    k_lstm<<<1, 1024, 0, stream>>>(feat, done, h0, c0, wihb, whhb, bsum,
                                   G, hsbuf, outHT, outCT);

    // --- heads ---
    k_head<<<64, 64, 0, stream>>>(hsbuf, headw, ba, bc, out);
}
